// LSTMModel_17265768530815
// MI455X (gfx1250) — compile-verified, matrix operand order checked by reading, not run on hardware
//
#include <hip/hip_runtime.h>
#include <stdint.h>

// ---------- CDNA5 WMMA types ----------
typedef __attribute__((ext_vector_type(16))) _Float16 v16h;
typedef __attribute__((ext_vector_type(8)))  _Float16 v8h;
typedef __attribute__((ext_vector_type(8)))  float    v8f;

__device__ __forceinline__ v8f wmma16(v16h a, v16h b, v8f c) {
  // D = A(16x32 f16) * B(32x16 f16) + C(16x16 f32)
  return __builtin_amdgcn_wmma_f32_16x16x32_f16(
      /*neg_a=*/false, a, /*neg_b=*/false, b,
      /*c_mod=*/(short)0, c, /*reuse_a=*/false, /*reuse_b=*/false);
}

__device__ __forceinline__ float sigm(float x) {
  return 1.0f / (1.0f + __expf(-x));
}
__device__ __forceinline__ float tanh_fast(float x) {
  return 1.0f - 2.0f / (__expf(2.0f * x) + 1.0f);
}

static constexpr int BATCH = 256;
static constexpr int TLEN  = 1024;

// K index held by (lane, element e) for 16-bit WMMA A and B fragments
// (identical K map per ISA 7.12.2): k = e + (e&8) + (laneHi ? 8 : 0)
__device__ __forceinline__ int kmap(int e, int hi) {
  return e + (e & 8) + (hi << 3);
}

// ---------- helper kernels ----------
__global__ void f32_to_f16_kernel(const float* __restrict__ in,
                                  _Float16* __restrict__ out, size_t n) {
  size_t i = (size_t)blockIdx.x * blockDim.x + threadIdx.x;
  size_t stride = (size_t)gridDim.x * blockDim.x;
  for (; i < n; i += stride) out[i] = (_Float16)in[i];
}

__global__ void zero_u32_kernel(uint32_t* __restrict__ p, size_t nwords) {
  size_t i = (size_t)blockIdx.x * blockDim.x + threadIdx.x;
  size_t stride = (size_t)gridDim.x * blockDim.x;
  for (; i < nwords; i += stride) p[i] = 0u;
}

// Pre-pack a weight matrix W [krows][4*U] (fp32, gate-concatenated columns)
// into WMMA B-fragment lane order, f16, one 16-col strip per (gate, ntile):
//   out[(((g*NT + nt)*KT + kt)*32 + lane)*16 + e]
__global__ void prepack_bfrags_kernel(const float* __restrict__ W, int krows,
                                      int U, int NT, int KT,
                                      _Float16* __restrict__ out) {
  int total = 4 * NT * KT * 512;
  for (int idx = blockIdx.x * blockDim.x + threadIdx.x; idx < total;
       idx += gridDim.x * blockDim.x) {
    int e    = idx & 15;
    int l    = (idx >> 4) & 31;
    int frag = idx >> 9;
    int kt   = frag % KT;
    int rest = frag / KT;
    int nt   = rest % NT;
    int g    = rest / NT;
    int hi   = l >> 4;
    int k    = kmap(e, hi) + kt * 32;
    int col  = nt * 16 + (l & 15);
    float v = (k < krows && col < U) ? W[(size_t)k * (4 * U) + g * U + col]
                                     : 0.0f;
    out[idx] = (_Float16)v;
  }
}

// ---------- fused LSTM layer scan ----------
// One workgroup per 16-row batch tile; 8 waves; wave w owns 16-column strip w
// (for ALL 4 gates). Wh fragments + bias fragments live in REGISTERS
// (loop-invariant), Wx fragments in LDS (copied once), h tile ping-pongs in
// LDS (1 barrier/step), c in registers. x_t A-frags software-pipelined.
template <int DINP, int U, int NT, int KXT, int KHT, int STOREP, bool LAST>
__global__ __launch_bounds__(256, 1) void lstm_scan_kernel(
    const _Float16* __restrict__ xin,   // [BATCH][TLEN][DINP] f16
    const _Float16* __restrict__ wxf,   // packed B-frags for Wx (global)
    const _Float16* __restrict__ whf,   // packed B-frags for Wh (global)
    const float* __restrict__ bias,     // [4*U]
    _Float16* __restrict__ xout,        // [BATCH][TLEN][STOREP] f16 (if !LAST)
    float* __restrict__ hfin)           // [BATCH][32] f32 (if LAST)
{
  constexpr int KHP = KHT * 32;             // padded hidden stride in LDS
  constexpr int HSZ = 16 * KHP;             // halves per h buffer
  constexpr int WXH = 4 * NT * KXT * 512;   // Wx frag halves held in LDS
  extern __shared__ __align__(16) _Float16 smem[];
  _Float16* wxlds = smem;                   // [WXH]
  _Float16* hbuf0 = smem + WXH;             // [HSZ]  h at even t
  _Float16* hbuf1 = smem + WXH + HSZ;       // [HSZ]  h at odd t

  const int tid  = threadIdx.x;
  const int wave = tid >> 5;
  const int lane = tid & 31;
  const int nrow = lane & 15;   // A row / BCD column within tile
  const int hi   = lane >> 4;
  const int hi8  = hi * 8;
  const int b0   = blockIdx.x * 16;

  // stage Wx fragments into LDS (16B vectors), zero both h buffers
  {
    const uint4* src = (const uint4*)wxf;
    uint4* dst = (uint4*)wxlds;
    for (int i = tid; i < WXH / 8; i += 256) dst[i] = src[i];
  }
  for (int i = tid; i < 2 * HSZ; i += 256) hbuf0[i] = (_Float16)0;
  __syncthreads();

  const bool compute = (wave < NT);
  const int  nt      = wave;

  // ---- loop-invariant state in registers ----
  v8f c;
#pragma unroll
  for (int r = 0; r < 8; ++r) c[r] = 0.0f;

  v8f  bacc[4];                 // bias pre-broadcast into C-fragment layout
  v16h whr[KHT][4];             // Wh B-fragments, resident in VGPRs
  v16h ax[KXT];                 // pipelined x_t A-fragments
  const _Float16* paxbase = xin + (size_t)(b0 + nrow) * TLEN * DINP + hi8;

  if (compute) {
    int col = nt * 16 + nrow;
#pragma unroll
    for (int g = 0; g < 4; ++g) {
      float bv = (col < U) ? bias[g * U + col] : 0.0f;
#pragma unroll
      for (int r = 0; r < 8; ++r) bacc[g][r] = bv;
    }
#pragma unroll
    for (int kt = 0; kt < KHT; ++kt) {
#pragma unroll
      for (int g = 0; g < 4; ++g) {
        whr[kt][g] = *(const v16h*)(whf +
            ((((size_t)g * NT + nt) * KHT + kt) * 32 + lane) * 16);
      }
    }
    // preload x A-fragments for t = 0
#pragma unroll
    for (int kt = 0; kt < KXT; ++kt) {
      ((v8h*)&ax[kt])[0] = *(const v8h*)(paxbase + kt * 32);
      ((v8h*)&ax[kt])[1] = *(const v8h*)(paxbase + kt * 32 + 16);
    }
  }

  auto step = [&](int t, const _Float16* __restrict__ hcur,
                  _Float16* __restrict__ hnxt) {
    v8f acc[4];
    if (compute) {
      // ---- x_t @ Wx (A from pipeline regs, B from LDS, C seeded by bias) ----
#pragma unroll
      for (int kt = 0; kt < KXT; ++kt) {
#pragma unroll
        for (int g = 0; g < 4; ++g) {
          v16h bf;
          const _Float16* pb = wxlds +
              ((((g * NT + nt) * KXT + kt) * 32) + lane) * 16;
          ((v8h*)&bf)[0] = *(const v8h*)(pb);
          ((v8h*)&bf)[1] = *(const v8h*)(pb + 8);
          acc[g] = wmma16(ax[kt], bf, (kt == 0) ? bacc[g] : acc[g]);
        }
      }
      // ---- issue x loads for t+1 (hidden behind Wh phase + gate math).
      // One-row-past-the-end read at t=TLEN-1 stays inside the workspace
      // for every layer's buffer placement; the value is never used.
      {
        const _Float16* pn = paxbase + (size_t)(t + 1) * DINP;
#pragma unroll
        for (int kt = 0; kt < KXT; ++kt) {
          ((v8h*)&ax[kt])[0] = *(const v8h*)(pn + kt * 32);
          ((v8h*)&ax[kt])[1] = *(const v8h*)(pn + kt * 32 + 16);
        }
      }
      // ---- h_{t-1} @ Wh (A from LDS ping buffer, B from registers) ----
#pragma unroll
      for (int kt = 0; kt < KHT; ++kt) {
        v16h ah;
        const _Float16* pah = hcur + nrow * KHP + kt * 32 + hi8;
        ((v8h*)&ah)[0] = *(const v8h*)(pah);
        ((v8h*)&ah)[1] = *(const v8h*)(pah + 16);
#pragma unroll
        for (int g = 0; g < 4; ++g) {
          acc[g] = wmma16(ah, whr[kt][g], acc[g]);
        }
      }
      // ---- gates, state update, write h into pong buffer ----
#pragma unroll
      for (int r = 0; r < 8; ++r) {
        float ig = sigm(acc[0][r]);
        float fg = sigm(acc[1][r]);
        float cb = tanh_fast(acc[2][r]);
        float og = sigm(acc[3][r]);
        float cn = fg * c[r] + ig * cb;
        c[r] = cn;
        float h  = og * tanh_fast(cn);
        int mrow = r + hi8;
        int col  = nt * 16 + nrow;
        hnxt[mrow * KHP + col] = (_Float16)h;
        if constexpr (!LAST) {
          xout[((size_t)(b0 + mrow) * TLEN + t) * STOREP + col] = (_Float16)h;
        }
      }
    }
    __syncthreads();   // hnxt complete -> becomes hcur of step t+1
  };

  for (int t = 0; t < TLEN; t += 2) {
    step(t,     hbuf0, hbuf1);
    step(t + 1, hbuf1, hbuf0);
  }

  if constexpr (LAST) {
    // final h lives in hbuf0 (TLEN even)
    if (compute) {
#pragma unroll
      for (int r = 0; r < 8; ++r) {
        int mrow = r + hi8;
        int col  = nt * 16 + nrow;
        hfin[(b0 + mrow) * 32 + col] = (float)hbuf0[mrow * KHP + col];
      }
    }
  }
}

// ---------- final dense [256,30] @ [30,1] + b ----------
__global__ void dense_out_kernel(const float* __restrict__ hf,
                                 const float* __restrict__ Wo,
                                 const float* __restrict__ bo,
                                 float* __restrict__ out) {
  int b = blockIdx.x * blockDim.x + threadIdx.x;
  if (b < BATCH) {
    float s = bo[0];
#pragma unroll
    for (int k = 0; k < 30; ++k) s += hf[b * 32 + k] * Wo[k];
    out[b] = s;
  }
}

// ---------- workspace layout ----------
static constexpr size_t OFF_B0 = 0;                         // seq f16 [256][1024][64]
static constexpr size_t SZ_B0  = (size_t)BATCH * TLEN * 64 * 2;
static constexpr size_t OFF_B1 = OFF_B0 + SZ_B0;            // L0 out [..][128]
static constexpr size_t SZ_B1  = (size_t)BATCH * TLEN * 128 * 2;
static constexpr size_t OFF_B2 = OFF_B1 + SZ_B1;            // L1 out [..][96]
static constexpr size_t SZ_B2  = (size_t)BATCH * TLEN * 96 * 2;
static constexpr size_t OFF_WX = OFF_B2 + SZ_B2;            // packed Wx frags
static constexpr size_t OFF_WH = OFF_WX + 131072;           // packed Wh frags
static constexpr size_t OFF_HF = OFF_WH + 131072;           // final h [256][32] f32

// dynamic LDS bytes per layer: Wx frags + 2x ping-pong h tile
static constexpr size_t smem_bytes(int NT, int KXT, int KHT) {
  return (size_t)(4 * NT * KXT * 512 + 2 * 16 * KHT * 32) * 2;
}

extern "C" void kernel_launch(void* const* d_in, const int* in_sizes, int n_in,
                              void* d_out, int out_size, void* d_ws,
                              size_t ws_size, hipStream_t stream) {
  (void)in_sizes; (void)n_in; (void)out_size; (void)ws_size;
  const float* seq  = (const float*)d_in[0];
  const float* Wx0  = (const float*)d_in[1];
  const float* Wh0  = (const float*)d_in[2];
  const float* b0   = (const float*)d_in[3];
  const float* Wx1  = (const float*)d_in[4];
  const float* Wh1  = (const float*)d_in[5];
  const float* b1   = (const float*)d_in[6];
  const float* Wx2  = (const float*)d_in[7];
  const float* Wh2  = (const float*)d_in[8];
  const float* b2   = (const float*)d_in[9];
  const float* Wx3  = (const float*)d_in[10];
  const float* Wh3  = (const float*)d_in[11];
  const float* b3   = (const float*)d_in[12];
  const float* Wout = (const float*)d_in[13];
  const float* bout = (const float*)d_in[14];

  char* ws = (char*)d_ws;
  _Float16* buf0 = (_Float16*)(ws + OFF_B0);
  _Float16* buf1 = (_Float16*)(ws + OFF_B1);
  _Float16* buf2 = (_Float16*)(ws + OFF_B2);
  _Float16* wxp  = (_Float16*)(ws + OFF_WX);
  _Float16* whp  = (_Float16*)(ws + OFF_WH);
  float*    hfin = (float*)(ws + OFF_HF);

  // sequence fp32 -> f16
  f32_to_f16_kernel<<<2048, 256, 0, stream>>>(seq, buf0,
                                              (size_t)BATCH * TLEN * 64);
  // zero padded-column regions of buf1+buf2 (contiguous)
  zero_u32_kernel<<<2048, 256, 0, stream>>>((uint32_t*)buf1,
                                            (SZ_B1 + SZ_B2) / 4);

  // ---- Layer 0: din=64(pad64), u=100 (N pad 112 -> 7 tiles, K pad 128) ----
  prepack_bfrags_kernel<<<224, 256, 0, stream>>>(Wx0, 64, 100, 7, 2, wxp);
  prepack_bfrags_kernel<<<224, 256, 0, stream>>>(Wh0, 100, 100, 7, 4, whp);
  lstm_scan_kernel<64, 100, 7, 2, 4, 128, false>
      <<<16, 256, smem_bytes(7, 2, 4), stream>>>(buf0, wxp, whp, b0, buf1,
                                                 nullptr);

  // ---- Layer 1: din=100(pad128), u=80 (5 tiles, K pad 96) ----
  prepack_bfrags_kernel<<<224, 256, 0, stream>>>(Wx1, 100, 80, 5, 4, wxp);
  prepack_bfrags_kernel<<<224, 256, 0, stream>>>(Wh1, 80, 80, 5, 3, whp);
  lstm_scan_kernel<128, 80, 5, 4, 3, 96, false>
      <<<16, 256, smem_bytes(5, 4, 3), stream>>>(buf1, wxp, whp, b1, buf2,
                                                 nullptr);

  // ---- Layer 2: din=80(pad96), u=50 (4 tiles, K pad 64); out reuses buf0 ----
  prepack_bfrags_kernel<<<224, 256, 0, stream>>>(Wx2, 80, 50, 4, 3, wxp);
  prepack_bfrags_kernel<<<224, 256, 0, stream>>>(Wh2, 50, 50, 4, 2, whp);
  lstm_scan_kernel<96, 50, 4, 3, 2, 64, false>
      <<<16, 256, smem_bytes(4, 3, 2), stream>>>(buf2, wxp, whp, b2, buf0,
                                                 nullptr);

  // ---- Layer 3: din=50(pad64), u=30 (2 tiles, K pad 32); final state only ----
  prepack_bfrags_kernel<<<224, 256, 0, stream>>>(Wx3, 50, 30, 2, 2, wxp);
  prepack_bfrags_kernel<<<224, 256, 0, stream>>>(Wh3, 30, 30, 2, 1, whp);
  lstm_scan_kernel<64, 30, 2, 2, 1, 64, true>
      <<<16, 256, smem_bytes(2, 2, 1), stream>>>(buf0, wxp, whp, b3, nullptr,
                                                 hfin);

  // ---- output head ----
  dense_out_kernel<<<1, 256, 0, stream>>>(hfin, Wout, bout, (float*)d_out);
}
